// InputUnitVisual_59304908423734
// MI455X (gfx1250) — matrix operand order repaired and checked by reading, not compiled, not run on hardware
//
#include <hip/hip_runtime.h>
#include <cstdint>
#include <cstddef>
#include <vector>

// ---------------------------------------------------------------------------
// Types for CDNA5 WMMA (wave32, v_wmma_f32_16x16x32_bf16)
// ---------------------------------------------------------------------------
typedef unsigned short bfu;                                   // bf16 storage
typedef __bf16  bf16_t;
typedef bf16_t  v16bf  __attribute__((ext_vector_type(16)));
typedef bf16_t  bf16x8 __attribute__((ext_vector_type(8)));
typedef float   v8f    __attribute__((ext_vector_type(8)));

__device__ __forceinline__ bfu f2bf(float f) {
  union { float f; uint32_t u; } v; v.f = f;
  uint32_t u = v.u;
  return (bfu)((u + 0x7FFFu + ((u >> 16) & 1u)) >> 16);      // RNE
}
__device__ __forceinline__ float bf2f(bfu b) {
  union { uint32_t u; float f; } v; v.u = ((uint32_t)b) << 16; return v.f;
}
__device__ __forceinline__ float sigm_(float x) { return 1.f / (1.f + expf(-x)); }
__device__ __forceinline__ float elu_(float x)  { return x > 0.f ? x : (expf(x) - 1.f); }

#define SHUF16(lo, hi) __builtin_shufflevector(lo, hi, 0,1,2,3,4,5,6,7,8,9,10,11,12,13,14,15)

// ---------------------------------------------------------------------------
// WMMA GEMM:  Out[M,N] = A[M,K](bf16) @ W[N,K]^T(bf16) + bias (+Cadd), act
// One wave per 16(M) x 64(N) tile: one A fragment reused across 4 B fragments.
// All fragments for a K-step are loaded BEFORE any WMMA issues so the loads
// form one clause with a single wait, then 4 back-to-back v_wmma.
// A fragment: lane = M-row(lane&15), K chunk  sel*16..+15
// B fragment: lane = N-row of W,    K chunk  sel*16..+15   (contiguous k, same order as A)
// ---------------------------------------------------------------------------
struct GemmArgs {
  const bfu* A; int lda;
  const bfu* W;                     // ldw == K
  const float* bias;                // nullable (len N)
  const float* Cadd; int ldcadd;    // nullable f32 addend
  float* Of; int ldof;              // nullable f32 out
  bfu*   Ob; int ldob;              // nullable bf16 out
  int M, N, K;
  int act;                          // 0=none 1=elu
};

__global__ __launch_bounds__(32) void gemm_bf16(GemmArgs g) {
  const int lane = threadIdx.x;
  const int n0 = blockIdx.x * 64;
  const int m0 = blockIdx.y * 16;
  const int r16 = lane & 15;
  const int sel = lane >> 4;
  const bfu* pa  = g.A + (size_t)(m0 + r16) * g.lda + sel * 16;
  const bfu* pw0 = g.W + (size_t)(n0 + r16) * g.K + sel * 16;
  const size_t wstride = (size_t)16 * g.K;
  v8f acc[4] = {{}, {}, {}, {}};
  for (int k = 0; k < g.K; k += 32) {
    // issue ALL loads first (one clause, one wait), then 4 dense WMMAs
    bf16x8 a0 = *reinterpret_cast<const bf16x8*>(pa + k);
    bf16x8 a1 = *reinterpret_cast<const bf16x8*>(pa + k + 8);
    bf16x8 blo[4], bhi[4];
#pragma unroll
    for (int t = 0; t < 4; ++t) {
      const bfu* pw = pw0 + (size_t)t * wstride;
      blo[t] = *reinterpret_cast<const bf16x8*>(pw + k);
      bhi[t] = *reinterpret_cast<const bf16x8*>(pw + k + 8);
    }
    v16bf av = SHUF16(a0, a1);
#pragma unroll
    for (int t = 0; t < 4; ++t) {
      v16bf bv = SHUF16(blo[t], bhi[t]);
      acc[t] = __builtin_amdgcn_wmma_f32_16x16x32_bf16(false, av, false, bv,
                                                       (short)0, acc[t], false, false);
    }
  }
  const int col = lane & 15;
  const int rb  = sel * 8;
#pragma unroll
  for (int t = 0; t < 4; ++t) {
    const int nn = n0 + t * 16 + col;
    float bia = g.bias ? g.bias[nn] : 0.f;
    for (int r = 0; r < 8; ++r) {
      int mm = m0 + rb + r;
      float v = acc[t][r] + bia;
      if (g.Cadd) v += g.Cadd[(size_t)mm * g.ldcadd + nn];
      if (g.act == 1) v = elu_(v);
      if (g.Of) g.Of[(size_t)mm * g.ldof + nn] = v;
      if (g.Ob) g.Ob[(size_t)mm * g.ldob + nn] = f2bf(v);
    }
  }
}

// Gated GEMM: out = elu(A@W^T + b) * sigmoid(A@Wg^T + bg).
// One wave per 16(M) x 32(N) tile; one A fragment feeds 4 WMMAs
// (2 n-tiles x {main, gate}); all loads grouped before the WMMA burst.
struct GemmGArgs {
  const bfu* A; int lda;
  const bfu* W; const bfu* Wg;
  const float* bias; const float* biasg;
  float* Of; int ldof;              // nullable
  bfu*   Ob; int ldob;              // nullable
  int M, N, K;
};

__global__ __launch_bounds__(32) void gemm_bf16_gated(GemmGArgs g) {
  const int lane = threadIdx.x;
  const int n0 = blockIdx.x * 32;
  const int m0 = blockIdx.y * 16;
  const int r16 = lane & 15;
  const int sel = lane >> 4;
  const bfu* pa   = g.A  + (size_t)(m0 + r16) * g.lda + sel * 16;
  const bfu* pw0  = g.W  + (size_t)(n0 + r16) * g.K + sel * 16;
  const bfu* pwg0 = g.Wg + (size_t)(n0 + r16) * g.K + sel * 16;
  const size_t wstride = (size_t)16 * g.K;
  v8f accM[2] = {{}, {}};
  v8f accG[2] = {{}, {}};
  for (int k = 0; k < g.K; k += 32) {
    bf16x8 a0 = *reinterpret_cast<const bf16x8*>(pa + k);
    bf16x8 a1 = *reinterpret_cast<const bf16x8*>(pa + k + 8);
    bf16x8 blo[2], bhi[2], clo[2], chi[2];
#pragma unroll
    for (int t = 0; t < 2; ++t) {
      const bfu* pw  = pw0  + (size_t)t * wstride;
      const bfu* pwg = pwg0 + (size_t)t * wstride;
      blo[t] = *reinterpret_cast<const bf16x8*>(pw + k);
      bhi[t] = *reinterpret_cast<const bf16x8*>(pw + k + 8);
      clo[t] = *reinterpret_cast<const bf16x8*>(pwg + k);
      chi[t] = *reinterpret_cast<const bf16x8*>(pwg + k + 8);
    }
    v16bf av = SHUF16(a0, a1);
#pragma unroll
    for (int t = 0; t < 2; ++t) {
      v16bf bv = SHUF16(blo[t], bhi[t]);
      accM[t] = __builtin_amdgcn_wmma_f32_16x16x32_bf16(false, av, false, bv,
                                                        (short)0, accM[t], false, false);
      v16bf cv = SHUF16(clo[t], chi[t]);
      accG[t] = __builtin_amdgcn_wmma_f32_16x16x32_bf16(false, av, false, cv,
                                                        (short)0, accG[t], false, false);
    }
  }
  const int col = lane & 15;
  const int rb  = sel * 8;
#pragma unroll
  for (int t = 0; t < 2; ++t) {
    const int nn = n0 + t * 16 + col;
    float bm = g.bias[nn], bg = g.biasg[nn];
    for (int r = 0; r < 8; ++r) {
      int mm = m0 + rb + r;
      float v = elu_(accM[t][r] + bm) * sigm_(accG[t][r] + bg);
      if (g.Of) g.Of[(size_t)mm * g.ldof + nn] = v;
      if (g.Ob) g.Ob[(size_t)mm * g.ldob + nn] = f2bf(v);
    }
  }
}

// ---------------------------------------------------------------------------
// Pointwise kernels
// ---------------------------------------------------------------------------
__global__ void k_f32_to_bf16(const float* s, bfu* d, int n) {
  int i = blockIdx.x * blockDim.x + threadIdx.x;
  if (i < n) d[i] = f2bf(s[i]);
}
__global__ void k_fill_zero_u32(uint32_t* p, int n) {
  int i = blockIdx.x * blockDim.x + threadIdx.x;
  if (i < n) p[i] = 0u;
}
// appearance (B,C,Fr=16,V=2048): mean over frames -> bf16 (B*C, 2048)
__global__ void k_app_mean(const float* x, bfu* out, int total) {
  int i = blockIdx.x * blockDim.x + threadIdx.x;
  if (i >= total) return;
  int v = i & 2047, bc = i >> 11;
  const float* p = x + (size_t)bc * 16 * 2048 + v;
  float s = 0.f;
  for (int f = 0; f < 16; ++f) s += p[(size_t)f * 2048];
  out[i] = f2bf(s * (1.f / 16.f));
}
// motion (B,C=8,V=2048): mean over clips -> bf16 (B,2048)
__global__ void k_mot_mean(const float* x, bfu* out, int total) {
  int i = blockIdx.x * blockDim.x + threadIdx.x;
  if (i >= total) return;
  int v = i & 2047, b = i >> 11;
  const float* p = x + (size_t)b * 8 * 2048 + v;
  float s = 0.f;
  for (int c = 0; c < 8; ++c) s += p[(size_t)c * 2048];
  out[i] = f2bf(s * (1.f / 8.f));
}
__global__ void k_bias_sum(const float* a, const float* b, float* o, int n) {
  int i = blockIdx.x * blockDim.x + threadIdx.x;
  if (i < n) o[i] = a[i] + b[i];
}
// LSTM cell: gates (128,2048) f32, torch order i,f,g,o; updates c, writes h bf16
__global__ void k_lstm_cell(const float* gates, float* c, bfu* hbf, int total) {
  int i = blockIdx.x * blockDim.x + threadIdx.x;
  if (i >= total) return;
  int d = i & 511, b = i >> 9;
  const float* gr = gates + (size_t)b * 2048;
  float ig = sigm_(gr[d]), fg = sigm_(gr[512 + d]);
  float gg = tanhf(gr[1024 + d]), og = sigm_(gr[1536 + d]);
  float cn = fg * c[i] + ig * gg;
  c[i] = cn;
  hbf[i] = f2bf(og * tanhf(cn));
}
// h1[(k,b),0:512]=ag[b*8+k], [512:1024]=mg[b]   (total 8*128*1024)
__global__ void k_concat_h1(const bfu* agb, const bfu* mgb, bfu* h1, int total) {
  int i = blockIdx.x * blockDim.x + threadIdx.x;
  if (i >= total) return;
  int d = i & 1023, b = (i >> 10) & 127, k = i >> 17;
  h1[i] = (d < 512) ? agb[(size_t)(b * 8 + k) * 512 + d]
                    : mgb[(size_t)b * 512 + (d - 512)];
}
struct Mask8 { unsigned int m[8]; };
// h2 stage for one j: rows (k,b); [0:512]=mean over masked m-objects; [512:]=q_proj
// m objects stored (k*128+b, o*512+d) with ld 7168
__global__ void k_concat_qc(const bfu* mobj, const bfu* qpb, bfu* h2, Mask8 mk, int total) {
  int i = blockIdx.x * blockDim.x + threadIdx.x;
  if (i >= total) return;
  int d = i & 1023, b = (i >> 10) & 127, k = i >> 17;
  if (d < 512) {
    unsigned mask = mk.m[k];
    const bfu* base = mobj + (size_t)(k * 128 + b) * 7168 + d;
    float s = 0.f;
    for (int o = 0; o < 14; ++o) if (mask & (1u << o)) s += bf2f(base[(size_t)o * 512]);
    h2[i] = f2bf(s / (float)__popc(mask));
  } else {
    h2[i] = qpb[(size_t)b * 512 + (d - 512)];
  }
}
// hv stage for one i: rows r=b*12+j; clip stored ((j*8+k)*128+b, d) ld 512
__global__ void k_concat_vm(const bfu* clip, const bfu* vmotb, bfu* hv, unsigned mask, int total) {
  int i = blockIdx.x * blockDim.x + threadIdx.x;
  if (i >= total) return;
  int d = i & 1023, r = i >> 10, b = r / 12, j = r % 12;
  if (d < 512) {
    float s = 0.f;
    for (int k = 0; k < 8; ++k)
      if (mask & (1u << k)) s += bf2f(clip[((size_t)(j * 8 + k) * 128 + b) * 512 + d]);
    hv[i] = f2bf(s / (float)__popc(mask));
  } else {
    hv[i] = vmotb[(size_t)b * 512 + (d - 512)];
  }
}
// hq stage for one t: rows r=b*12+j; vm stored (ii*1536+r, d) ld 512
__global__ void k_concat_vq(const bfu* vmb, const bfu* qpb, bfu* hq, unsigned mask, int total) {
  int i = blockIdx.x * blockDim.x + threadIdx.x;
  if (i >= total) return;
  int d = i & 1023, r = i >> 10, b = r / 12;
  if (d < 512) {
    float s = 0.f;
    for (int ii = 0; ii < 6; ++ii)
      if (mask & (1u << ii)) s += bf2f(vmb[((size_t)ii * 1536 + r) * 512 + d]);
    hq[i] = f2bf(s / (float)__popc(mask));
  } else {
    hq[i] = qpb[(size_t)b * 512 + (d - 512)];
  }
}
// final permute: vqf (t, r=b*12+j, d) -> out (b, t*12+j, d)
__global__ void k_final(const float* vqf, float* out, int total) {
  int i = blockIdx.x * blockDim.x + threadIdx.x;
  if (i >= total) return;
  int d = i & 511, r = (i >> 9) % 1536, t = i / (1536 * 512);
  int b = r / 12, j = r % 12;
  out[((size_t)b * 48 + t * 12 + j) * 512 + d] = vqf[i];
}

// ---------------------------------------------------------------------------
// Host: numpy-compatible RandomState(0) (MT19937 + legacy shuffle/choice)
// ---------------------------------------------------------------------------
namespace nprng {
struct MT {
  uint32_t mt[624]; int mti;
  void seed(uint32_t s) {
    for (int i = 0; i < 624; ++i) { mt[i] = s; s = 1812433253u * (s ^ (s >> 30)) + (uint32_t)i + 1u; }
    mti = 624;
  }
  uint32_t next() {
    if (mti >= 624) {
      for (int kk = 0; kk < 624; ++kk) {
        uint32_t y = (mt[kk] & 0x80000000u) | (mt[(kk + 1) % 624] & 0x7fffffffu);
        mt[kk] = mt[(kk + 397) % 624] ^ (y >> 1) ^ ((y & 1u) ? 0x9908b0dfu : 0u);
      }
      mti = 0;
    }
    uint32_t y = mt[mti++];
    y ^= y >> 11; y ^= (y << 7) & 0x9d2c5680u; y ^= (y << 15) & 0xefc60000u; y ^= y >> 18;
    return y;
  }
  uint64_t interval(uint64_t mx) {           // uniform on [0, mx], rejection
    if (mx == 0) return 0;
    uint64_t mask = mx;
    mask |= mask >> 1; mask |= mask >> 2; mask |= mask >> 4;
    mask |= mask >> 8; mask |= mask >> 16; mask |= mask >> 32;
    uint64_t v;
    while ((v = ((uint64_t)next() & mask)) > mx) {}
    return v;
  }
  long choice1(long n, std::vector<int>& scratch) { // permutation(n)[0]
    scratch.resize((size_t)n);
    for (long i = 0; i < n; ++i) scratch[(size_t)i] = (int)i;
    for (long i = n - 1; i >= 1; --i) {
      long j = (long)interval((uint64_t)i);
      int t = scratch[(size_t)i]; scratch[(size_t)i] = scratch[(size_t)j]; scratch[(size_t)j] = t;
    }
    return scratch[0];
  }
};
} // namespace nprng

static void unrank_comb(const long C[17][17], int n, int r, long rank, unsigned* mask_out) {
  unsigned mask = 0; int c = 0;
  for (int slot = 0; slot < r; ++slot) {
    for (;;) {
      if (c >= n) { break; }
      long cnt = C[n - 1 - c][r - 1 - slot];
      if (rank < cnt) { mask |= 1u << c; ++c; break; }
      rank -= cnt; ++c;
    }
  }
  *mask_out = mask;
}

// ---------------------------------------------------------------------------
// Orchestration
// ---------------------------------------------------------------------------
extern "C" void kernel_launch(void* const* d_in, const int* in_sizes, int n_in,
                              void* d_out, int out_size, void* d_ws, size_t ws_size,
                              hipStream_t stream) {
  (void)in_sizes; (void)n_in; (void)out_size; (void)ws_size;
  // ---- inputs (f32) ----
  const float* app   = (const float*)d_in[0];   // (128,8,16,2048)
  const float* mot   = (const float*)d_in[1];   // (128,8,2048)
  const float* qemb  = (const float*)d_in[2];   // (128,512)
  const float* Wq = (const float*)d_in[3];  const float* bq = (const float*)d_in[4];
  const float* Wm = (const float*)d_in[5];  const float* bm = (const float*)d_in[6];
  const float* Wa = (const float*)d_in[7];  const float* ba = (const float*)d_in[8];
  const float* Wvm = (const float*)d_in[9]; const float* bvm = (const float*)d_in[10];
  const float* Wag1 = (const float*)d_in[11]; const float* bag1 = (const float*)d_in[12];
  const float* Wag2 = (const float*)d_in[13]; const float* bag2 = (const float*)d_in[14];
  const float* Wmg1 = (const float*)d_in[15]; const float* bmg1 = (const float*)d_in[16];
  const float* Wmg2 = (const float*)d_in[17]; const float* bmg2 = (const float*)d_in[18];
  const float* Wih = (const float*)d_in[19]; const float* Whh = (const float*)d_in[20];
  const float* bih = (const float*)d_in[21]; const float* bhh = (const float*)d_in[22];
  const float* Wcm = (const float*)d_in[23]; const float* bcm = (const float*)d_in[24];
  const float* Wcq = (const float*)d_in[25]; const float* bcq = (const float*)d_in[26];
  const float* Wcqg = (const float*)d_in[27]; const float* bcqg = (const float*)d_in[28];
  const float* Wvmc = (const float*)d_in[29]; const float* bvmc = (const float*)d_in[30];
  const float* Wvq = (const float*)d_in[31]; const float* bvq = (const float*)d_in[32];
  const float* Wvqg = (const float*)d_in[33]; const float* bvqg = (const float*)d_in[34];
  float* out = (float*)d_out;

  // ---- workspace bump allocator ----
  size_t off = 0;
  auto wsa = [&](size_t bytes) -> void* {
    void* p = (void*)((char*)d_ws + off);
    off += (bytes + 255) & ~(size_t)255;
    return p;
  };
  bfu* appmean = (bfu*)wsa((size_t)1024 * 2048 * 2);
  bfu* motmean = (bfu*)wsa((size_t)128 * 2048 * 2);
  bfu* motbf   = (bfu*)wsa((size_t)1024 * 2048 * 2);
  bfu* qembbf  = (bfu*)wsa((size_t)128 * 512 * 2);
  bfu* Wq_b  = (bfu*)wsa((size_t)512 * 512 * 2);
  bfu* Wm_b  = (bfu*)wsa((size_t)512 * 2048 * 2);
  bfu* Wa_b  = (bfu*)wsa((size_t)512 * 2048 * 2);
  bfu* Wvm_b = (bfu*)wsa((size_t)512 * 512 * 2);
  bfu* Wag1_b = (bfu*)wsa((size_t)512 * 512 * 2);
  bfu* Wag2_b = (bfu*)wsa((size_t)512 * 512 * 2);
  bfu* Wmg1_b = (bfu*)wsa((size_t)512 * 512 * 2);
  bfu* Wmg2_b = (bfu*)wsa((size_t)512 * 512 * 2);
  bfu* Wih_b = (bfu*)wsa((size_t)2048 * 2048 * 2);
  bfu* Whh_b = (bfu*)wsa((size_t)2048 * 512 * 2);
  bfu* Wcm_b = (bfu*)wsa((size_t)15 * 512 * 1024 * 2);
  float* bsum  = (float*)wsa((size_t)2048 * 4);
  float* xW    = (float*)wsa((size_t)1024 * 2048 * 4);
  float* gates = (float*)wsa((size_t)128 * 2048 * 4);
  bfu*   hbf   = (bfu*)wsa((size_t)128 * 512 * 2);
  float* cst   = (float*)wsa((size_t)128 * 512 * 4);
  bfu* agin = (bfu*)wsa((size_t)1024 * 512 * 2);
  bfu* t1   = (bfu*)wsa((size_t)1024 * 512 * 2);
  bfu* agb  = (bfu*)wsa((size_t)1024 * 512 * 2);
  bfu* mgin = (bfu*)wsa((size_t)128 * 512 * 2);
  bfu* t2   = (bfu*)wsa((size_t)128 * 512 * 2);
  bfu* mgb  = (bfu*)wsa((size_t)128 * 512 * 2);
  bfu* qpb  = (bfu*)wsa((size_t)128 * 512 * 2);
  bfu* vmotb = (bfu*)wsa((size_t)128 * 512 * 2);
  bfu* h1   = (bfu*)wsa((size_t)1024 * 1024 * 2);
  bfu* mobj = (bfu*)wsa((size_t)1024 * 7168 * 2);
  bfu* wst1 = (bfu*)wsa((size_t)512 * 1024 * 2);
  bfu* wst2 = (bfu*)wsa((size_t)512 * 1024 * 2);
  bfu* h2st = (bfu*)wsa((size_t)1024 * 1024 * 2);
  bfu* clip = (bfu*)wsa((size_t)12 * 1024 * 512 * 2);
  bfu* hvst = (bfu*)wsa((size_t)1536 * 1024 * 2);
  bfu* vmb  = (bfu*)wsa((size_t)6 * 1536 * 512 * 2);
  bfu* hqst = (bfu*)wsa((size_t)1536 * 1024 * 2);
  float* vqf = (float*)wsa((size_t)4 * 1536 * 512 * 4);

  // ---- reproduce np.random.RandomState(0) subset choices (host, pure) ----
  long C[17][17];
  for (int i = 0; i <= 16; ++i)
    for (int j = 0; j <= 16; ++j)
      C[i][j] = (j == 0) ? 1 : (j > i ? 0 : C[i - 1][j - 1] + C[i - 1][j]);
  nprng::MT rng; rng.seed(0u);
  std::vector<int> scratch; scratch.reserve(12870);
  unsigned qcMask[8][12], vmMask[6], vqMask[4];
  for (int k = 0; k < 8; ++k) {
    for (int sid = 1; sid <= 14; ++sid)                 // first CRN: objects identical,
      (void)rng.choice1(C[16][16 - sid], scratch);      // but RNG state must advance
    for (int sid = 1; sid <= 12; ++sid) {
      int scale = 14 - sid;
      long idx = rng.choice1(C[14][scale], scratch);
      unrank_comb(C, 14, scale, idx, &qcMask[k][sid - 1]);
    }
  }
  for (int sid = 1; sid <= 6; ++sid) {
    int scale = 8 - sid;
    long idx = rng.choice1(C[8][scale], scratch);
    unrank_comb(C, 8, scale, idx, &vmMask[sid - 1]);
  }
  for (int sid = 1; sid <= 4; ++sid) {
    int scale = 6 - sid;
    long idx = rng.choice1(C[6][scale], scratch);
    unrank_comb(C, 6, scale, idx, &vqMask[sid - 1]);
  }

  // ---- launch helpers ----
  auto cdiv = [](int a, int b) { return (a + b - 1) / b; };
  auto conv = [&](const float* s, bfu* d, int n) {
    k_f32_to_bf16<<<cdiv(n, 256), 256, 0, stream>>>(s, d, n);
  };
  auto gemm = [&](const bfu* A, int lda, const bfu* W, const float* bias,
                  const float* Cadd, int ldcadd, float* Of, int ldof,
                  bfu* Ob, int ldob, int M, int N, int K, int act) {
    GemmArgs g{A, lda, W, bias, Cadd, ldcadd, Of, ldof, Ob, ldob, M, N, K, act};
    gemm_bf16<<<dim3(N / 64, M / 16), 32, 0, stream>>>(g);
  };
  auto gemmg = [&](const bfu* A, int lda, const bfu* W, const bfu* Wg,
                   const float* bias, const float* biasg, float* Of, int ldof,
                   bfu* Ob, int ldob, int M, int N, int K) {
    GemmGArgs g{A, lda, W, Wg, bias, biasg, Of, ldof, Ob, ldob, M, N, K};
    gemm_bf16_gated<<<dim3(N / 32, M / 16), 32, 0, stream>>>(g);
  };

  // ---- stage 0: conversions & means ----
  conv(Wq, Wq_b, 512 * 512);       conv(Wm, Wm_b, 512 * 2048);
  conv(Wa, Wa_b, 512 * 2048);      conv(Wvm, Wvm_b, 512 * 512);
  conv(Wag1, Wag1_b, 512 * 512);   conv(Wag2, Wag2_b, 512 * 512);
  conv(Wmg1, Wmg1_b, 512 * 512);   conv(Wmg2, Wmg2_b, 512 * 512);
  conv(Wih, Wih_b, 2048 * 2048);   conv(Whh, Whh_b, 2048 * 512);
  conv(Wcm, Wcm_b, 15 * 512 * 1024);
  conv(qemb, qembbf, 128 * 512);   conv(mot, motbf, 1024 * 2048);
  k_app_mean<<<cdiv(1024 * 2048, 256), 256, 0, stream>>>(app, appmean, 1024 * 2048);
  k_mot_mean<<<cdiv(128 * 2048, 256), 256, 0, stream>>>(mot, motmean, 128 * 2048);
  k_bias_sum<<<cdiv(2048, 256), 256, 0, stream>>>(bih, bhh, bsum, 2048);

  // ---- projections + collapsed GCN chains (mean commutes with linear) ----
  gemm(appmean, 2048, Wa_b, ba, nullptr, 0, nullptr, 0, agin, 512, 1024, 512, 2048, 0);
  gemm(agin, 512, Wag1_b, bag1, nullptr, 0, nullptr, 0, t1,   512, 1024, 512, 512, 0);
  gemm(t1,   512, Wag2_b, bag2, nullptr, 0, nullptr, 0, agb,  512, 1024, 512, 512, 0);
  gemm(motmean, 2048, Wm_b, bm, nullptr, 0, nullptr, 0, mgin, 512, 128, 512, 2048, 0);
  gemm(mgin, 512, Wmg1_b, bmg1, nullptr, 0, nullptr, 0, t2,  512, 128, 512, 512, 0);
  gemm(t2,   512, Wmg2_b, bmg2, nullptr, 0, nullptr, 0, mgb, 512, 128, 512, 512, 0);
  gemm(qembbf, 512, Wq_b, bq, nullptr, 0, nullptr, 0, qpb, 512, 128, 512, 512, 0);

  // ---- LSTM over motion (T=8): precompute x@W_ih^T, then recurrent steps ----
  gemm(motbf, 2048, Wih_b, bsum, nullptr, 0, xW, 2048, nullptr, 0, 1024, 2048, 2048, 0);
  k_fill_zero_u32<<<cdiv(128 * 512 / 2, 256), 256, 0, stream>>>((uint32_t*)hbf, 128 * 512 / 2);
  k_fill_zero_u32<<<cdiv(128 * 512, 256), 256, 0, stream>>>((uint32_t*)cst, 128 * 512);
  for (int t = 0; t < 8; ++t) {
    gemm(hbf, 512, Whh_b, nullptr, xW + (size_t)t * 2048, 8 * 2048,
         gates, 2048, nullptr, 0, 128, 2048, 512, 0);
    k_lstm_cell<<<cdiv(128 * 512, 256), 256, 0, stream>>>(gates, cst, hbf, 128 * 512);
  }
  gemm(hbf, 512, Wvm_b, bvm, nullptr, 0, nullptr, 0, vmotb, 512, 128, 512, 512, 0);

  // ---- CRN1: identical objects => one stacked GEMM over Wcm[1..14] ----
  k_concat_h1<<<cdiv(1024 * 1024, 256), 256, 0, stream>>>(agb, mgb, h1, 1024 * 1024);
  gemm(h1, 1024, Wcm_b + (size_t)512 * 1024, bcm + 512, nullptr, 0,
       nullptr, 0, mobj, 7168, 1024, 7168, 1024, 1);

  // ---- CRN2 (gated, per sid j=0..11): subset-mean + concat -> gated GEMM ----
  for (int j = 0; j < 12; ++j) {
    conv(Wcq + (size_t)(j + 1) * 512 * 1024, wst1, 512 * 1024);
    conv(Wcqg + (size_t)(j + 1) * 512 * 1024, wst2, 512 * 1024);
    Mask8 mk;
    for (int k = 0; k < 8; ++k) mk.m[k] = qcMask[k][j];
    k_concat_qc<<<cdiv(1024 * 1024, 256), 256, 0, stream>>>(mobj, qpb, h2st, mk, 1024 * 1024);
    gemmg(h2st, 1024, wst1, wst2, bcq + (size_t)(j + 1) * 512, bcqg + (size_t)(j + 1) * 512,
          nullptr, 0, clip + (size_t)j * 1024 * 512, 512, 1024, 512, 1024);
  }

  // ---- video-motion CRN (i=0..5) ----
  for (int i = 0; i < 6; ++i) {
    conv(Wvmc + (size_t)(i + 1) * 512 * 1024, wst1, 512 * 1024);
    k_concat_vm<<<cdiv(1536 * 1024, 256), 256, 0, stream>>>(clip, vmotb, hvst, vmMask[i], 1536 * 1024);
    gemm(hvst, 1024, wst1, bvmc + (size_t)(i + 1) * 512, nullptr, 0,
         nullptr, 0, vmb + (size_t)i * 1536 * 512, 512, 1536, 512, 1024, 1);
  }

  // ---- video-question CRN (gated, t=0..3) ----
  for (int t = 0; t < 4; ++t) {
    conv(Wvq + (size_t)(t + 1) * 512 * 1024, wst1, 512 * 1024);
    conv(Wvqg + (size_t)(t + 1) * 512 * 1024, wst2, 512 * 1024);
    k_concat_vq<<<cdiv(1536 * 1024, 256), 256, 0, stream>>>(vmb, qpb, hqst, vqMask[t], 1536 * 1024);
    gemmg(hqst, 1024, wst1, wst2, bvq + (size_t)(t + 1) * 512, bvqg + (size_t)(t + 1) * 512,
          vqf + (size_t)t * 1536 * 512, 512, nullptr, 0, 1536, 512, 1024);
  }

  // ---- final permute (t, b*12+j, d) -> (b, t*12+j, d) ----
  k_final<<<cdiv(4 * 1536 * 512, 256), 256, 0, stream>>>(vqf, out, 4 * 1536 * 512);
}
